// Decoder_28630251995819
// MI455X (gfx1250) — compile-verified, matrix operand order checked by reading, not run on hardware
//
#include <hip/hip_runtime.h>
#include <hip/hip_bf16.h>
#include <math.h>

// ---------------------------------------------------------------------------
// Decoder pipeline for MI455X (gfx1250, wave32, WMMA).
//   x  = sin(0.01*(latent@lin0_w.T+b))
//   x  = sin(0.01*conv(x|ps0, g0, conv0))
//   x  = onera_interp(x, pos0->pos1)
//   x  = sin(0.01*conv(x|ps1, g1, conv0))
//   x  = onera_interp(x, pos1->pos2)
//   x  = sin(0.01*conv(x|ps2, g2, conv1))
//   out= x@lin1_w.T + lin1_b
// ---------------------------------------------------------------------------

typedef __attribute__((ext_vector_type(16))) _Float16 v16h;
typedef __attribute__((ext_vector_type(8)))  float    v8f;

#define WAVES     8          // 256 threads/block = 8 wave32
#define INV_PI    0.3183098861837907f
#define OMEGA_K   0.1f
#define OMEGA_DEC 0.01f
#define TANPI6    0.57735026918962576f
#define SCALE_C   0.78571428571428571f
#define YREF      1.1963f

#if defined(__has_builtin)
#  if __has_builtin(__builtin_amdgcn_permlane16)
#    define USE_PERMLANE16 1
#  endif
#endif

// K index held in half j of the A/B operand for this lane-half (ISA 7.12.2)
__device__ __forceinline__ int wmmaK(int j, int hi) {
    return (j & 7) + ((j >> 3) << 4) + hi * 8;   // {0..7,16..23} (+8 for hi lanes)
}

// xor-butterfly within each 16-lane half: VALU v_permlane16_b32 when
// available (no LDS port traffic), ds_bpermute fallback otherwise.
__device__ __forceinline__ float xor16(float v, int off, unsigned s0, unsigned s1) {
#ifdef USE_PERMLANE16
    (void)off;
    unsigned u = __float_as_uint(v);
    return __uint_as_float(__builtin_amdgcn_permlane16(u, u, s0, s1, false, false));
#else
    (void)s0; (void)s1;
    return __shfl_xor(v, off, 32);
#endif
}

// ------------------------------ tiny kernels -------------------------------

__global__ void decoderKernel(const float* __restrict__ latent,
                              const float* __restrict__ w,
                              const float* __restrict__ b,
                              float* __restrict__ out, int n) {
    int t = blockIdx.x * blockDim.x + threadIdx.x;
    if (t >= n * 32) return;
    int node = t >> 5, h = t & 31;
    float acc = b[h];
    #pragma unroll 8
    for (int k = 0; k < 32; ++k) acc += latent[node * 32 + k] * w[h * 32 + k];
    out[t] = __sinf(OMEGA_DEC * acc);
}

__global__ void sinBiasKernel(const float* __restrict__ agg,
                              const float* __restrict__ bias,
                              float* __restrict__ out, int total) {
    int t = blockIdx.x * blockDim.x + threadIdx.x;
    if (t >= total) return;
    out[t] = __sinf(OMEGA_DEC * (agg[t] + bias[t & 31]));
}

__global__ void transformKernel(const float* __restrict__ in,
                                float* __restrict__ out, int n) {
    int t = blockIdx.x * blockDim.x + threadIdx.x;
    if (t >= n) return;
    float x = in[t * 3], y = in[t * 3 + 1], z = in[t * 3 + 2];
    float nx = x - TANPI6 * y;
    float s  = 1.f + SCALE_C * (y / YREF);
    out[t * 3] = nx * s; out[t * 3 + 1] = y * s; out[t * 3 + 2] = z * s;
}

__global__ void headKernel(const float* __restrict__ x,
                           const float* __restrict__ w,
                           const float* __restrict__ b,
                           float* __restrict__ out, int n) {
    int t = blockIdx.x * blockDim.x + threadIdx.x;
    if (t >= n * 5) return;
    int node = t / 5, o = t % 5;
    float acc = b[o];
    #pragma unroll 8
    for (int h = 0; h < 32; ++h) acc += x[node * 32 + h] * w[o * 32 + h];
    out[t] = acc;
}

// ------------------------- graph kernel conv (WMMA) ------------------------
// One wave owns a tile of 16 edges.
//  v-tile  : v = x_j @ W2 via two v_wmma (pos-scale tail folded into C) --
//            output lands directly in the C-layout registers the epilogue
//            consumes (no LDS round trip).
//  channel : A = sin(0.1*(base + c*wc)) packed in registers (A-layout K
//            order), two v_wmma against W1^T with b1 pre-folded into C,
//            epilogue sin + dot with v, permlane16 butterfly over k, one
//            f32 atomic per (edge, channel) = segment_sum.

__global__ void __launch_bounds__(256, 1)
convKernel(const float* __restrict__ xin,   // [N,32] node features
           const float* __restrict__ ps,    // [N,3]  pos_scale (pos AND extra chans)
           const int*   __restrict__ ei,    // [2,E]
           int E, int N,
           const float* __restrict__ W0, const float* __restrict__ W1,
           const float* __restrict__ W2, const float* __restrict__ b0,
           const float* __restrict__ b1, const float* __restrict__ b2,
           float* __restrict__ agg)         // [N,32] zero-initialized
{
    __shared__ float sW0[32 * 3];
    __shared__ float sWc[32];
    __shared__ float sB0[32];
    __shared__ float sW1[32 * 32];
    __shared__ float sB1[32];
    __shared__ float sW2[35 * 32];
    __shared__ float sB2[35];
    __shared__ float sPs [WAVES][16][3];
    __shared__ float sCt [WAVES][16];
    __shared__ int   sDst[WAVES][16];

    const int tid = threadIdx.x;
    for (int i = tid; i < 32 * 4; i += 256) {
        float v = W0[i]; int h = i >> 2, d = i & 3;
        if (d == 3) sWc[h] = v; else sW0[h * 3 + d] = v;
    }
    for (int i = tid; i < 32 * 32; i += 256) sW1[i] = W1[i];
    for (int i = tid; i < 35 * 32; i += 256) sW2[i] = W2[i];
    if (tid < 32) { sB0[tid] = b0[tid]; sB1[tid] = b1[tid]; }
    if (tid < 35) sB2[tid] = b2[tid];
    __syncthreads();

    const int lane = tid & 31, wv = tid >> 5;
    const int hi = lane >> 4, n = lane & 15;

    // B operands for h1 = h0 @ W1^T : B[k][col] = W1[col][k]
    v16h BW1a, BW1b;
    // B operands for v = x_j @ W2   : B[k][col] = W2[k][col]
    v16h BW2a, BW2b;
    #pragma unroll
    for (int j = 0; j < 16; ++j) {
        int k = wmmaK(j, hi);
        BW1a[j] = (_Float16)sW1[n * 32 + k];
        BW1b[j] = (_Float16)sW1[(n + 16) * 32 + k];
        BW2a[j] = (_Float16)sW2[k * 32 + n];
        BW2b[j] = (_Float16)sW2[k * 32 + n + 16];
    }
    // C operand pre-loaded with b1 (C[r][col] = b1[col])
    v8f cb1a, cb1b;
    #pragma unroll
    for (int r = 0; r < 8; ++r) { cb1a[r] = sB1[n]; cb1b[r] = sB1[n + 16]; }

    const long totalE = (long)E + (long)N;        // + self loops
    const int tiles = (int)((totalE + 15) >> 4);
    const int wavesTotal = gridDim.x * WAVES;
    const int groups = (tiles + wavesTotal - 1) / wavesTotal;

    for (int g = 0; g < groups; ++g) {
        const int tile = g * wavesTotal + blockIdx.x * WAVES + wv;
        const bool inTile = tile < tiles;
        const long e = (long)tile * 16 + n;       // this lane's edge (m == n)
        const bool valid = inTile && (e < totalE);

        int src = 0, dst = -1;
        if (valid) {
            if (e < (long)E) {
                src = ei[e]; dst = ei[(long)E + e];
                __builtin_prefetch(ei + e + 512, 0, 1);
            } else {
                src = dst = (int)(e - E);         // self loop
            }
        }
        float sx = 0.f, sy = 0.f, sz = 0.f, rx = 0.f, ry = 0.f, rz = 0.f;
        if (valid) {
            int s3 = src * 3, d3 = dst * 3;
            sx = ps[s3]; sy = ps[s3 + 1]; sz = ps[s3 + 2];
            rx = ps[d3] - sx; ry = ps[d3 + 1] - sy; rz = ps[d3 + 2] - sz;
        }
        float rho = sqrtf(rx * rx + ry * ry + rz * rz);
        float th = 0.f, ph = 0.f;
        if (rho > 0.f) { th = atan2f(ry, rx) * INV_PI; ph = asinf(rz / rho) * INV_PI; }

        // base/wc directly in registers, already in A-layout K order + 0.1x
        float bR[16], wR[16];
        #pragma unroll
        for (int j = 0; j < 16; ++j) {
            int h = wmmaK(j, hi);
            float base = rho * sW0[h * 3] + th * sW0[h * 3 + 1] +
                         ph * sW0[h * 3 + 2] + sB0[h];
            bR[j] = OMEGA_K * base;
            wR[j] = OMEGA_K * sWc[h];
        }

        // x_j row (b128 loads) + cterm; A operand for the v-tile GEMM
        float xj[32];
        float ct = 0.f;
        if (valid) {
            const float4* xr4 = (const float4*)(xin + (long)src * 32);
            #pragma unroll
            for (int i4 = 0; i4 < 8; ++i4) {
                float4 q = xr4[i4];
                xj[4 * i4 + 0] = q.x; xj[4 * i4 + 1] = q.y;
                xj[4 * i4 + 2] = q.z; xj[4 * i4 + 3] = q.w;
            }
            #pragma unroll
            for (int i = 0; i < 32; ++i) ct += xj[i] * sB2[i];
            ct += sx * sB2[32] + sy * sB2[33] + sz * sB2[34];
        } else {
            #pragma unroll
            for (int i = 0; i < 32; ++i) xj[i] = 0.f;
        }
        v16h AX;
        #pragma unroll
        for (int j = 0; j < 16; ++j) AX[j] = (_Float16)xj[wmmaK(j, hi)];

        if (inTile && hi == 0) {
            sPs[wv][n][0] = sx; sPs[wv][n][1] = sy; sPs[wv][n][2] = sz;
            sCt[wv][n] = ct;
            sDst[wv][n] = dst;
        }
        __syncthreads();

        if (inTile) {
            // v-tile: C = pos-scale tail (K = 32..34), then v = AX@W2 + C
            v8f cv0, cv1;
            #pragma unroll
            for (int r = 0; r < 8; ++r) {
                int me = r + hi * 8;
                float p0 = sPs[wv][me][0], p1 = sPs[wv][me][1], p2 = sPs[wv][me][2];
                cv0[r] = p0 * sW2[32 * 32 + n]      + p1 * sW2[33 * 32 + n]      + p2 * sW2[34 * 32 + n];
                cv1[r] = p0 * sW2[32 * 32 + n + 16] + p1 * sW2[33 * 32 + n + 16] + p2 * sW2[34 * 32 + n + 16];
            }
            v8f vA = __builtin_amdgcn_wmma_f32_16x16x32_f16(
                false, AX, false, BW2a, (short)0, cv0, false, false);
            v8f vB = __builtin_amdgcn_wmma_f32_16x16x32_f16(
                false, AX, false, BW2b, (short)0, cv1, false, false);

            float ctW = 0.f; int dW = -1;
            if (n < 8) { int me = n + 8 * hi; ctW = sCt[wv][me]; dW = sDst[wv][me]; }

            for (int c = 0; c < 32; ++c) {
                const float cf = (float)c;
                v16h A;
                #pragma unroll
                for (int j = 0; j < 16; ++j)
                    A[j] = (_Float16)__sinf(fmaf(cf, wR[j], bR[j]));

                v8f acc0 = __builtin_amdgcn_wmma_f32_16x16x32_f16(
                    false, A, false, BW1a, (short)0, cb1a, false, false);
                v8f acc1 = __builtin_amdgcn_wmma_f32_16x16x32_f16(
                    false, A, false, BW1b, (short)0, cb1b, false, false);

                float p[8];
                #pragma unroll
                for (int r = 0; r < 8; ++r) {
                    p[r] = __sinf(OMEGA_K * acc0[r]) * vA[r] +
                           __sinf(OMEGA_K * acc1[r]) * vB[r];
                }
                // butterfly sum over the 16 k-lanes of each half (VALU permlane)
                #pragma unroll
                for (int r = 0; r < 8; ++r) p[r] += xor16(p[r], 1, 0x67452301u, 0xEFCDAB89u);
                #pragma unroll
                for (int r = 0; r < 8; ++r) p[r] += xor16(p[r], 2, 0x54761032u, 0xDCFE98BAu);
                #pragma unroll
                for (int r = 0; r < 8; ++r) p[r] += xor16(p[r], 4, 0x32107654u, 0xBA98FEDCu);
                #pragma unroll
                for (int r = 0; r < 8; ++r) p[r] += xor16(p[r], 8, 0xFEDCBA98u, 0x76543210u);

                #pragma unroll
                for (int r = 0; r < 8; ++r) {
                    if (n == r && dW >= 0)
                        atomicAdd(&agg[(long)dW * 32 + c], p[r] + ctW);
                }
            }
        }
        __syncthreads();
    }
}

// ----------------------------- knn interpolate -----------------------------
// Brute-force k=3 with inverse-squared-distance weights; per-query onera mask
// selects transformed vs raw coordinates. Reference points tiled through LDS
// (both coordinate sets, b128 staging) and amortized over 32 queries/block.

#define KCH 2048
#define QPW 4

__global__ void __launch_bounds__(256, 1)
knnInterpKernel(const float* __restrict__ f,     // [Nx,32]
                const float* __restrict__ px, const float* __restrict__ tpx, int Nx,
                const float* __restrict__ py, const float* __restrict__ tpy, int Ny,
                float* __restrict__ out)         // [Ny,32]
{
    __shared__ __align__(16) float sP0[KCH * 3];
    __shared__ __align__(16) float sP1[KCH * 3];
    __shared__ float sMd[WAVES][96];
    __shared__ int   sMi[WAVES][96];

    const int tid = threadIdx.x, lane = tid & 31, wv = tid >> 5;
    const int qBase = blockIdx.x * 32 + wv * QPW;

    float qx[QPW], qy[QPW], qz[QPW];
    bool  qm[QPW], qv[QPW];
    float bd[QPW][3]; int bi[QPW][3];
    #pragma unroll
    for (int q = 0; q < QPW; ++q) {
        int qi = qBase + q;
        qv[q] = qi < Ny;
        int qq = qv[q] ? qi : 0;
        bool msk = py[qq * 3 + 1] < YREF;
        qm[q] = msk;
        const float* s = msk ? tpy : py;
        qx[q] = s[qq * 3]; qy[q] = s[qq * 3 + 1]; qz[q] = s[qq * 3 + 2];
        bd[q][0] = bd[q][1] = bd[q][2] = 3.0e38f;
        bi[q][0] = bi[q][1] = bi[q][2] = 0;
    }

    const int nChunks = (Nx + KCH - 1) / KCH;
    for (int ch = 0; ch < nChunks; ++ch) {
        const int base = ch * KCH;
        const int limF = Nx * 3 - base * 3;       // valid floats in this chunk
        __syncthreads();
        for (int i = tid; i < KCH * 3 / 4; i += 256) {
            int fo = i * 4;
            float4 a0, a1;
            if (fo + 4 <= limF) {
                a0 = ((const float4*)(px  + (size_t)base * 3))[i];
                a1 = ((const float4*)(tpx + (size_t)base * 3))[i];
            } else {
                float t0[4], t1[4];
                #pragma unroll
                for (int k = 0; k < 4; ++k) {
                    int fidx = fo + k;
                    bool ok = fidx < limF;
                    t0[k] = ok ? px [(size_t)base * 3 + fidx] : 1e30f;
                    t1[k] = ok ? tpx[(size_t)base * 3 + fidx] : 1e30f;
                }
                a0 = make_float4(t0[0], t0[1], t0[2], t0[3]);
                a1 = make_float4(t1[0], t1[1], t1[2], t1[3]);
            }
            ((float4*)sP0)[i] = a0;
            ((float4*)sP1)[i] = a1;
        }
        __syncthreads();
        for (int t = lane; t < KCH; t += 32) {
            float ax0 = sP0[t * 3], ay0 = sP0[t * 3 + 1], az0 = sP0[t * 3 + 2];
            float ax1 = sP1[t * 3], ay1 = sP1[t * 3 + 1], az1 = sP1[t * 3 + 2];
            int gi = base + t;
            #pragma unroll
            for (int q = 0; q < QPW; ++q) {
                float ax = qm[q] ? ax1 : ax0;
                float ay = qm[q] ? ay1 : ay0;
                float az = qm[q] ? az1 : az0;
                float dx = ax - qx[q], dy = ay - qy[q], dz = az - qz[q];
                float d2 = dx * dx + dy * dy + dz * dz;
                if (d2 < bd[q][2]) {
                    if (d2 < bd[q][0]) {
                        bd[q][2] = bd[q][1]; bi[q][2] = bi[q][1];
                        bd[q][1] = bd[q][0]; bi[q][1] = bi[q][0];
                        bd[q][0] = d2;       bi[q][0] = gi;
                    } else if (d2 < bd[q][1]) {
                        bd[q][2] = bd[q][1]; bi[q][2] = bi[q][1];
                        bd[q][1] = d2;       bi[q][1] = gi;
                    } else {
                        bd[q][2] = d2;       bi[q][2] = gi;
                    }
                }
            }
        }
    }

    // merge 32 lanes x top-3 -> global top-3 (redundantly in every lane)
    for (int q = 0; q < QPW; ++q) {
        __syncthreads();
        #pragma unroll
        for (int j = 0; j < 3; ++j) {
            sMd[wv][lane * 3 + j] = bd[q][j];
            sMi[wv][lane * 3 + j] = bi[q][j];
        }
        __syncthreads();
        float td0 = 3e38f, td1 = 3e38f, td2 = 3e38f;
        int   ti0 = 0,     ti1 = 0,     ti2 = 0;
        for (int j = 0; j < 96; ++j) {
            float d2 = sMd[wv][j]; int gi = sMi[wv][j];
            if (d2 < td2) {
                if (d2 < td0)      { td2 = td1; ti2 = ti1; td1 = td0; ti1 = ti0; td0 = d2; ti0 = gi; }
                else if (d2 < td1) { td2 = td1; ti2 = ti1; td1 = d2; ti1 = gi; }
                else               { td2 = d2; ti2 = gi; }
            }
        }
        int qi = qBase + q;
        if (qv[q]) {
            float w0 = 1.f / fmaxf(td0, 1e-16f);
            float w1 = 1.f / fmaxf(td1, 1e-16f);
            float w2 = 1.f / fmaxf(td2, 1e-16f);
            float inv = 1.f / (w0 + w1 + w2);
            float o = w0 * f[(long)ti0 * 32 + lane] +
                      w1 * f[(long)ti1 * 32 + lane] +
                      w2 * f[(long)ti2 * 32 + lane];
            out[(long)qi * 32 + lane] = o * inv;
        }
    }
}

// --------------------------------- driver ----------------------------------

extern "C" void kernel_launch(void* const* d_in, const int* in_sizes, int n_in,
                              void* d_out, int out_size, void* d_ws, size_t ws_size,
                              hipStream_t stream) {
    (void)in_sizes; (void)n_in; (void)out_size; (void)ws_size;

    // setup_inputs() dict order; conv dicts flattened in jax pytree (sorted-key)
    // order: W0, W1, W2, b0, b1, b2, bias.
    const float* latent = (const float*)d_in[0];
    const float* lin0_w = (const float*)d_in[1];
    const float* lin0_b = (const float*)d_in[2];
    const float* lin1_w = (const float*)d_in[3];
    const float* lin1_b = (const float*)d_in[4];
    const float* c0W0 = (const float*)d_in[5];
    const float* c0W1 = (const float*)d_in[6];
    const float* c0W2 = (const float*)d_in[7];
    const float* c0b0 = (const float*)d_in[8];
    const float* c0b1 = (const float*)d_in[9];
    const float* c0b2 = (const float*)d_in[10];
    const float* c0bias = (const float*)d_in[11];
    const float* c1W0 = (const float*)d_in[12];
    const float* c1W1 = (const float*)d_in[13];
    const float* c1W2 = (const float*)d_in[14];
    const float* c1b0 = (const float*)d_in[15];
    const float* c1b1 = (const float*)d_in[16];
    const float* c1b2 = (const float*)d_in[17];
    const float* c1bias = (const float*)d_in[18];
    const int* ei0 = (const int*)d_in[19];
    const int* ei1 = (const int*)d_in[20];
    const int* ei2 = (const int*)d_in[21];
    const float* ps0 = (const float*)d_in[22];
    const float* ps1 = (const float*)d_in[23];
    const float* ps2 = (const float*)d_in[24];
    const float* p0 = (const float*)d_in[25];
    const float* p1 = (const float*)d_in[26];
    const float* p2 = (const float*)d_in[27];

    const int N0 = 4096, N1 = 16384, N2 = 65536;
    const int E0 = 65536, E1 = 262144, E2 = 1048576;

    float* ws = (float*)d_ws;
    float* x0   = ws;                 // 4096*32
    float* agg0 = x0   + 131072;
    float* x1   = agg0 + 131072;
    float* xi1  = x1   + 131072;      // 16384*32
    float* agg1 = xi1  + 524288;
    float* x2   = agg1 + 524288;
    float* xi2  = x2   + 524288;      // 65536*32
    float* agg2 = xi2  + 2097152;
    float* x3   = agg2 + 2097152;
    float* tpx  = x3   + 2097152;     // up to 65536*3
    float* tpy  = tpx  + 196608;

    auto launchConv = [&](const float* xin, const float* ps, const int* ei,
                          int E, int N,
                          const float* W0, const float* W1, const float* W2,
                          const float* b0, const float* b1, const float* b2,
                          float* agg) {
        long totalE = (long)E + N;
        int tiles = (int)((totalE + 15) >> 4);
        int blocks = (tiles + WAVES - 1) / WAVES;
        if (blocks > 2048) blocks = 2048;
        hipMemsetAsync(agg, 0, (size_t)N * 32 * sizeof(float), stream);
        convKernel<<<blocks, 256, 0, stream>>>(xin, ps, ei, E, N,
                                               W0, W1, W2, b0, b1, b2, agg);
    };

    // Stage A: decoder lin0 + sin
    decoderKernel<<<(N0 * 32 + 255) / 256, 256, 0, stream>>>(latent, lin0_w, lin0_b, x0, N0);

    // Stage B: conv0 on graph 0
    launchConv(x0, ps0, ei0, E0, N0, c0W0, c0W1, c0W2, c0b0, c0b1, c0b2, agg0);
    sinBiasKernel<<<(N0 * 32 + 255) / 256, 256, 0, stream>>>(agg0, c0bias, x1, N0 * 32);

    // Stage C: onera interp pos0 -> pos1
    transformKernel<<<(N0 + 255) / 256, 256, 0, stream>>>(p0, tpx, N0);
    transformKernel<<<(N1 + 255) / 256, 256, 0, stream>>>(p1, tpy, N1);
    knnInterpKernel<<<(N1 + 31) / 32, 256, 0, stream>>>(x1, p0, tpx, N0, p1, tpy, N1, xi1);

    // Stage D: conv0 (reused) on graph 1
    launchConv(xi1, ps1, ei1, E1, N1, c0W0, c0W1, c0W2, c0b0, c0b1, c0b2, agg1);
    sinBiasKernel<<<(N1 * 32 + 255) / 256, 256, 0, stream>>>(agg1, c0bias, x2, N1 * 32);

    // Stage E: onera interp pos1 -> pos2
    transformKernel<<<(N1 + 255) / 256, 256, 0, stream>>>(p1, tpx, N1);
    transformKernel<<<(N2 + 255) / 256, 256, 0, stream>>>(p2, tpy, N2);
    knnInterpKernel<<<(N2 + 31) / 32, 256, 0, stream>>>(x2, p1, tpx, N1, p2, tpy, N2, xi2);

    // Stage F: conv1 on graph 2
    launchConv(xi2, ps2, ei2, E2, N2, c1W0, c1W1, c1W2, c1b0, c1b1, c1b2, agg2);
    sinBiasKernel<<<(N2 * 32 + 255) / 256, 256, 0, stream>>>(agg2, c1bias, x3, N2 * 32);

    // Stage G: head
    headKernel<<<(N2 * 5 + 255) / 256, 256, 0, stream>>>(x3, lin1_w, lin1_b, (float*)d_out, N2);
}
